// MultiheadAttention_11201274708414
// MI455X (gfx1250) — matrix operand order repaired and checked by reading, not compiled, not run on hardware
//
#include <hip/hip_runtime.h>
#include <hip/hip_bf16.h>

// MHA forward for B=2, S=2048, D_MODEL=1024, H=16, D_K=64 on gfx1250.
// v_wmma_f32_16x16x32_f16 everywhere; operands staged through LDS with
// CDNA5 async-to-LDS loads (double-buffered) to cut L2 traffic.

typedef __attribute__((ext_vector_type(16))) _Float16 v16h;
typedef __attribute__((ext_vector_type(8)))  _Float16 v8h;
typedef __attribute__((ext_vector_type(8)))  float    v8f;
typedef __attribute__((ext_vector_type(4)))  int      v4i;

#define DEV static __device__ __forceinline__

#if defined(__has_builtin)
#if __has_builtin(__builtin_amdgcn_global_load_async_to_lds_b128)
#define HAS_ASYNC_COPY 1
#endif
#endif

typedef __attribute__((address_space(1))) v4i* gv4p;  // global v4i*
typedef __attribute__((address_space(3))) v4i* lv4p;  // LDS v4i*

// copy 16 bytes global -> LDS (async on CDNA5, sync fallback otherwise)
DEV void copy16_g2l(const _Float16* g, _Float16* s) {
#if defined(HAS_ASYNC_COPY)
  __builtin_amdgcn_global_load_async_to_lds_b128(
      (gv4p)(v4i*)const_cast<_Float16*>(g), (lv4p)(v4i*)s, 0, 0);
#else
  *(v8h*)s = *(const v8h*)g;
#endif
}

DEV void wait_async_slab(bool keep_one_slab, int per_slab) {
#if defined(HAS_ASYNC_COPY)
  if (keep_one_slab) {
    if (per_slab == 4) asm volatile("s_wait_asynccnt 0x4" ::: "memory");
    else               asm volatile("s_wait_asynccnt 0x2" ::: "memory");
  } else {
    asm volatile("s_wait_asynccnt 0x0" ::: "memory");
  }
#else
  (void)keep_one_slab; (void)per_slab;
#endif
}

DEV v8f wmma_f16(v16h a, v16h b, v8f c) {
  return __builtin_amdgcn_wmma_f32_16x16x32_f16(
      false, a, false, b, (short)0, c, false, false);
}

// A-fragment (16x32 f16, MxK): lane l(0-15) = row l; hi half selects K phase.
// frag[0..7] = K = hi*8 + 0..7 ; frag[8..15] = K = 16 + hi*8 + 0..7
DEV v16h load_frag_a(const _Float16* base, int ld, int row0, int k0, int lane) {
  int l = lane & 15, hi = (lane >> 4) & 1;
  const _Float16* p = base + (size_t)(row0 + l) * ld + k0 + hi * 8;
  v8h a = *(const v8h*)p;
  v8h b = *(const v8h*)(p + 16);
  v16h f;
#pragma unroll
  for (int i = 0; i < 8; ++i) { f[i] = a[i]; f[i + 8] = b[i]; }
  return f;
}

// B-fragment (32x16 f16, KxN) from an NxK row-major source (B[k][n]=Src[n][k]):
// lane l(0-15) = column n; frag[0..7] = K = hi*8 + 0..7 ; frag[8..15] = K = 16 + hi*8 + 0..7
DEV v16h load_frag_b_nk(const _Float16* base, int ld, int n0, int k0, int lane) {
  int l = lane & 15, hi = (lane >> 4) & 1;
  const _Float16* p = base + (size_t)(n0 + l) * ld + k0 + hi * 8;
  v8h a = *(const v8h*)p;
  v8h b = *(const v8h*)(p + 16);
  v16h f;
#pragma unroll
  for (int i = 0; i < 8; ++i) { f[i] = a[i]; f[i + 8] = b[i]; }
  return f;
}

// ---------------- f32 -> f16 cast ----------------
__global__ void cast_f32_f16_kernel(const float* __restrict__ src,
                                    _Float16* __restrict__ dst, int n) {
  int i = blockIdx.x * blockDim.x + threadIdx.x;
  int stride = gridDim.x * blockDim.x;
  for (; i < n; i += stride) dst[i] = (_Float16)src[i];
}

// ---------------- GEMM: y = X(f16)[M,K] * W(f16)[N,K]^T + bias ----------------
// Block (8 waves) -> 128x128 tile; per 32-k step an A-slab (128x32) and W-slab
// (128x32) are async-staged into double-buffered LDS and shared by all waves.
// Each wave computes 32x64 (8 WMMAs per k-step).
// mode 0: f16 head-split [B,H,S,64]; mode 1: f16 transposed [B,H,64,S];
// mode 2: f32 plain [M,N].
__global__ __launch_bounds__(256) void gemm_nt_kernel(
    const _Float16* __restrict__ X, const _Float16* __restrict__ W,
    const float* __restrict__ bias, void* __restrict__ dst, int mode, int M) {
  const int K = 1024;
  __shared__ __align__(16) _Float16 As[2][128 * 32];  // 8 KB each
  __shared__ __align__(16) _Float16 Bs[2][128 * 32];
  int tid = threadIdx.x;
  int lane = tid & 31, wid = tid >> 5;
  int wm = wid & 3, wn = wid >> 2;             // 4 x 2 wave grid
  int bm = blockIdx.x >> 3, bn = blockIdx.x & 7;
  int m0 = bm * 128, n0 = bn * 128;

  // slab issue: 512 16B chunks per slab, 2 per thread per matrix
  auto issue = [&](int buf, int k0) {
#pragma unroll
    for (int i = 0; i < 2; ++i) {
      int c = tid + i * 256;
      int row = c >> 2, q = c & 3;             // 4 chunks per 32-half row
      copy16_g2l(&X[(size_t)(m0 + row) * K + k0 + q * 8], &As[buf][row * 32 + q * 8]);
    }
#pragma unroll
    for (int i = 0; i < 2; ++i) {
      int c = tid + i * 256;
      int row = c >> 2, q = c & 3;
      copy16_g2l(&W[(size_t)(n0 + row) * K + k0 + q * 8], &Bs[buf][row * 32 + q * 8]);
    }
  };

  v8f acc[2][4];
#pragma unroll
  for (int mi = 0; mi < 2; ++mi)
#pragma unroll
    for (int t = 0; t < 4; ++t) acc[mi][t] = v8f{};

  issue(0, 0);
  const int NK = K / 32;  // 32
  for (int ki = 0; ki < NK; ++ki) {
    int cur = ki & 1;
    bool hn = (ki + 1) < NK;
    if (hn) issue(cur ^ 1, (ki + 1) * 32);
    wait_async_slab(hn, 4);
    __syncthreads();  // slab[cur] visible to all waves

    v16h a0 = load_frag_a(&As[cur][0], 32, wm * 32,      0, lane);
    v16h a1 = load_frag_a(&As[cur][0], 32, wm * 32 + 16, 0, lane);
    v16h b0 = load_frag_b_nk(&Bs[cur][0], 32, wn * 64,      0, lane);
    v16h b1 = load_frag_b_nk(&Bs[cur][0], 32, wn * 64 + 16, 0, lane);
    v16h b2 = load_frag_b_nk(&Bs[cur][0], 32, wn * 64 + 32, 0, lane);
    v16h b3 = load_frag_b_nk(&Bs[cur][0], 32, wn * 64 + 48, 0, lane);
    acc[0][0] = wmma_f16(a0, b0, acc[0][0]);
    acc[0][1] = wmma_f16(a0, b1, acc[0][1]);
    acc[0][2] = wmma_f16(a0, b2, acc[0][2]);
    acc[0][3] = wmma_f16(a0, b3, acc[0][3]);
    acc[1][0] = wmma_f16(a1, b0, acc[1][0]);
    acc[1][1] = wmma_f16(a1, b1, acc[1][1]);
    acc[1][2] = wmma_f16(a1, b2, acc[1][2]);
    acc[1][3] = wmma_f16(a1, b3, acc[1][3]);

    __syncthreads();  // all reads of slab[cur] done before it is re-filled
  }

  int l = lane & 15, hi = (lane >> 4) & 1;
#pragma unroll
  for (int t = 0; t < 4; ++t) {
    int col = n0 + wn * 64 + 16 * t + l;  // 0..1023
    float bv = bias[col];
    int h = col >> 6, d = col & 63;
#pragma unroll
    for (int mi = 0; mi < 2; ++mi) {
#pragma unroll
      for (int r = 0; r < 8; ++r) {
        int row = m0 + wm * 32 + 16 * mi + r + 8 * hi;  // flattened b*S+s
        float v = acc[mi][t][r] + bv;
        int b = row >> 11, s = row & 2047;
        if (mode == 0) {
          ((_Float16*)dst)[(((size_t)(b * 16 + h)) * 2048 + s) * 64 + d] = (_Float16)v;
        } else if (mode == 1) {
          ((_Float16*)dst)[(((size_t)(b * 16 + h)) * 64 + d) * 2048 + s] = (_Float16)v;
        } else {
          ((float*)dst)[(size_t)row * 1024 + col] = v;
        }
      }
    }
  }
}

// ---------------- Flash attention ----------------
// Block (8 waves) = 8 query tiles of one (b,h). K/V chunks (32 keys) are
// async-staged into double-buffered LDS and shared by all 8 waves (8x cut in
// K/V traffic). Online softmax per wave; P re-laid-out through per-wave LDS.
__global__ __launch_bounds__(256) void attn_kernel(
    const _Float16* __restrict__ Q,   // [B*H, S, 64]
    const _Float16* __restrict__ Kt,  // [B*H, S, 64]
    const _Float16* __restrict__ Vt,  // [B*H, 64, S]
    _Float16* __restrict__ ctx) {     // [B*S, 1024]
  const int S = 2048, DK = 64;
  __shared__ __align__(16) _Float16 Ks[2][32 * 64];  // 4 KB each
  __shared__ __align__(16) _Float16 Vs[2][64 * 32];  // 4 KB each
  __shared__ __align__(16) _Float16 Ps[8][16 * 32];  // 1 KB per wave
  int tid = threadIdx.x;
  int lane = tid & 31, wib = tid >> 5;
  int bh = blockIdx.x >> 4;                         // 0..31, uniform per block
  int s0 = ((blockIdx.x & 15) * 8 + wib) * 16;      // query tile start
  const _Float16* Qb = Q  + (size_t)bh * S * DK;
  const _Float16* Kb = Kt + (size_t)bh * S * DK;
  const _Float16* Vb = Vt + (size_t)bh * DK * S;
  int l = lane & 15, hi = (lane >> 4) & 1;

  v16h qa0 = load_frag_a(Qb, DK, s0, 0, lane);
  v16h qa1 = load_frag_a(Qb, DK, s0, 32, lane);

  float mstate[8], lstate[8];
#pragma unroll
  for (int r = 0; r < 8; ++r) { mstate[r] = -1e30f; lstate[r] = 0.0f; }
  v8f o[4] = {v8f{}, v8f{}, v8f{}, v8f{}};
  const float scale = 0.125f;  // 1/sqrt(64)

  // chunk staging: K rows [kb..kb+31] x 64 dims; V dims [0..63] x 32 keys
  auto issue = [&](int buf, int kb) {
    { int row = tid >> 3, q = tid & 7;  // 32 rows x 8 chunks
      copy16_g2l(&Kb[(size_t)(kb + row) * DK + q * 8], &Ks[buf][row * 64 + q * 8]); }
    { int row = tid >> 2, q = tid & 3;  // 64 rows x 4 chunks
      copy16_g2l(&Vb[(size_t)row * S + kb + q * 8], &Vs[buf][row * 32 + q * 8]); }
  };

  issue(0, 0);
  const int NC = S / 32;  // 64 chunks
  for (int ci = 0; ci < NC; ++ci) {
    int cur = ci & 1;
    bool hn = (ci + 1) < NC;
    if (hn) issue(cur ^ 1, (ci + 1) * 32);
    wait_async_slab(hn, 2);
    __syncthreads();  // chunk[cur] visible

    v16h kf0 = load_frag_b_nk(&Ks[cur][0], 64, 0,  0,  lane);
    v16h kf1 = load_frag_b_nk(&Ks[cur][0], 64, 0,  32, lane);
    v16h kf2 = load_frag_b_nk(&Ks[cur][0], 64, 16, 0,  lane);
    v16h kf3 = load_frag_b_nk(&Ks[cur][0], 64, 16, 32, lane);
    v16h vf0 = load_frag_b_nk(&Vs[cur][0], 32, 0,  0, lane);
    v16h vf1 = load_frag_b_nk(&Vs[cur][0], 32, 16, 0, lane);
    v16h vf2 = load_frag_b_nk(&Vs[cur][0], 32, 32, 0, lane);
    v16h vf3 = load_frag_b_nk(&Vs[cur][0], 32, 48, 0, lane);

    // scores: two 16x16 tiles covering the 32 keys (DK=64 -> 2 WMMAs each)
    v8f t0 = {}, t1 = {};
    t0 = wmma_f16(qa0, kf0, t0);
    t0 = wmma_f16(qa1, kf1, t0);
    t1 = wmma_f16(qa0, kf2, t1);
    t1 = wmma_f16(qa1, kf3, t1);

    float corr[8];
#pragma unroll
    for (int r = 0; r < 8; ++r) {
      float a = t0[r] * scale, b = t1[r] * scale;
      float v = fmaxf(a, b);
#pragma unroll
      for (int off = 1; off < 16; off <<= 1) v = fmaxf(v, __shfl_xor(v, off, 32));
      float mn = fmaxf(mstate[r], v);
      corr[r] = __expf(mstate[r] - mn);
      float p0 = __expf(a - mn), p1 = __expf(b - mn);
      float rs = p0 + p1;
#pragma unroll
      for (int off = 1; off < 16; off <<= 1) rs += __shfl_xor(rs, off, 32);
      lstate[r] = lstate[r] * corr[r] + rs;
      mstate[r] = mn;
      int row = r + 8 * hi;
      Ps[wib][row * 32 + l]      = (_Float16)p0;
      Ps[wib][row * 32 + 16 + l] = (_Float16)p1;
    }
#pragma unroll
    for (int t = 0; t < 4; ++t)
#pragma unroll
      for (int r = 0; r < 8; ++r) o[t][r] *= corr[r];

    asm volatile("s_wait_dscnt 0" ::: "memory");  // RAW through LDS, same wave

    const _Float16* lp = &Ps[wib][l * 32 + hi * 8];
    v16h pa;
#pragma unroll
    for (int i = 0; i < 8; ++i) { pa[i] = lp[i]; pa[i + 8] = lp[16 + i]; }

    o[0] = wmma_f16(pa, vf0, o[0]);
    o[1] = wmma_f16(pa, vf1, o[1]);
    o[2] = wmma_f16(pa, vf2, o[2]);
    o[3] = wmma_f16(pa, vf3, o[3]);

    __syncthreads();  // all reads of chunk[cur] done before refill
  }

  float inv[8];
#pragma unroll
  for (int r = 0; r < 8; ++r) inv[r] = 1.0f / lstate[r];
  int b = bh >> 4, h = bh & 15;
#pragma unroll
  for (int t = 0; t < 4; ++t) {
    int col = h * 64 + 16 * t + l;
#pragma unroll
    for (int r = 0; r < 8; ++r) {
      int row = s0 + r + 8 * hi;
      ctx[((size_t)b * 2048 + row) * 1024 + col] = (_Float16)(o[t][r] * inv[r]);
    }
  }
}

extern "C" void kernel_launch(void* const* d_in, const int* in_sizes, int n_in,
                              void* d_out, int out_size, void* d_ws, size_t ws_size,
                              hipStream_t stream) {
  const float* q  = (const float*)d_in[0];
  const float* k  = (const float*)d_in[1];
  const float* v  = (const float*)d_in[2];
  const float* Wq = (const float*)d_in[3];
  const float* bq = (const float*)d_in[4];
  const float* Wk = (const float*)d_in[5];
  const float* bk = (const float*)d_in[6];
  const float* Wv = (const float*)d_in[7];
  const float* bv = (const float*)d_in[8];
  const float* Wo = (const float*)d_in[9];
  const float* bo = (const float*)d_in[10];

  const size_t MB = 1u << 20;
  char* ws = (char*)d_ws;
  _Float16* qh  = (_Float16*)(ws + 0 * MB);   // [4096,1024]
  _Float16* kh  = (_Float16*)(ws + 8 * MB);
  _Float16* vh  = (_Float16*)(ws + 16 * MB);
  _Float16* wqh = (_Float16*)(ws + 24 * MB);  // [1024,1024]
  _Float16* wkh = (_Float16*)(ws + 26 * MB);
  _Float16* wvh = (_Float16*)(ws + 28 * MB);
  _Float16* woh = (_Float16*)(ws + 30 * MB);
  _Float16* Qb  = (_Float16*)(ws + 32 * MB);  // [B,H,S,64]
  _Float16* Kb  = (_Float16*)(ws + 40 * MB);  // [B,H,S,64]
  _Float16* Vtb = (_Float16*)(ws + 48 * MB);  // [B,H,64,S]
  _Float16* ctx = (_Float16*)(ws + 56 * MB);  // [B*S,1024]

  const int NX = 4096 * 1024, NW = 1024 * 1024;
  cast_f32_f16_kernel<<<2048, 256, 0, stream>>>(q,  qh,  NX);
  cast_f32_f16_kernel<<<2048, 256, 0, stream>>>(k,  kh,  NX);
  cast_f32_f16_kernel<<<2048, 256, 0, stream>>>(v,  vh,  NX);
  cast_f32_f16_kernel<<<1024, 256, 0, stream>>>(Wq, wqh, NW);
  cast_f32_f16_kernel<<<1024, 256, 0, stream>>>(Wk, wkh, NW);
  cast_f32_f16_kernel<<<1024, 256, 0, stream>>>(Wv, wvh, NW);
  cast_f32_f16_kernel<<<1024, 256, 0, stream>>>(Wo, woh, NW);

  // projections: (4096/128)*(1024/128) = 32*8 = 256 blocks of 8 waves
  gemm_nt_kernel<<<256, 256, 0, stream>>>(qh, wqh, bq, (void*)Qb,  0, 4096);
  gemm_nt_kernel<<<256, 256, 0, stream>>>(kh, wkh, bk, (void*)Kb,  0, 4096);
  gemm_nt_kernel<<<256, 256, 0, stream>>>(vh, wvh, bv, (void*)Vtb, 1, 4096);

  // attention: 32 (b,h) * 16 blocks (8 q-tiles each) = 512 blocks
  attn_kernel<<<512, 256, 0, stream>>>(Qb, Kb, Vtb, ctx);

  // output projection -> f32 d_out
  gemm_nt_kernel<<<256, 256, 0, stream>>>(ctx, woh, bo, d_out, 2, 4096);
}
